// HypergraphConvolution_28174985462032
// MI455X (gfx1250) — matrix operand order repaired and checked by reading, not compiled, not run on hardware
//
#include <hip/hip_runtime.h>
#include <hip/hip_bf16.h>
#include <cmath>

typedef __attribute__((ext_vector_type(16))) _Float16 v16h;
typedef __attribute__((ext_vector_type(8)))  float    v8f;

#define B_  8
#define C_  128
#define N_  9216
#define E_  16
#define NTILES (N_/16)     // 576
#define NCH  72            // N_/128 chunks for hyper aggregation

// ---------------- WMMA operand loaders (CDNA5 16x16x32 f16 layouts) ----------

// A (16x32 f16): lane L -> M = L&15, half = L>>4.
// element i: K = (half<<4) + i
static __device__ __forceinline__ v16h load_A_h(const _Float16* p, int stride, int lane) {
  int m = lane & 15, half = lane >> 4;
  v16h a;
#pragma unroll
  for (int i = 0; i < 16; ++i) {
    int k = (half << 4) + i;
    a[i] = p[m * stride + k];
  }
  return a;
}

// A with only K<16 valid (K=16..31 zero padded): used for E=16 contractions
static __device__ __forceinline__ v16h load_A_pad16_f32(const float* p, int stride, int lane) {
  int m = lane & 15, half = lane >> 4;
  v16h a;
#pragma unroll
  for (int i = 0; i < 16; ++i) {
    int k = (half << 4) + i;
    a[i] = (k < 16) ? (_Float16)p[m * stride + k] : (_Float16)0.0f;
  }
  return a;
}

// B (32x16): lane L -> N = L&15, K group = (L>>4)*16; element i -> K = kg + i.
// B[k,n] = w[(n0+n)*ldw + k0 + k]  (row-major [out,in] f16 weight used transposed)
static __device__ __forceinline__ v16h load_B_wT_h(const _Float16* w, int ldw, int n0, int k0, int lane) {
  int n = n0 + (lane & 15);
  int kg = (lane >> 4) << 4;
  v16h b;
#pragma unroll
  for (int i = 0; i < 16; ++i) b[i] = w[(size_t)n * ldw + k0 + kg + i];
  return b;
}

// B[k,n] = p[k*stride + n0+n], rows valid for k<kmax (else zero)
static __device__ __forceinline__ v16h load_B_rm_h(const _Float16* p, int stride, int n0, int kmax, int lane) {
  int n = n0 + (lane & 15);
  int kg = (lane >> 4) << 4;
  v16h b;
#pragma unroll
  for (int i = 0; i < 16; ++i) {
    int k = kg + i;
    b[i] = (k < kmax) ? p[k * stride + n] : (_Float16)0.0f;
  }
  return b;
}

static __device__ __forceinline__ v8f wmma_f16(v16h a, v16h b, v8f c) {
  return __builtin_amdgcn_wmma_f32_16x16x32_f16(false, a, false, b, (short)0, c, false, false);
}

static __device__ __forceinline__ float gelu_f(float x) {
  return 0.5f * x * (1.0f + erff(x * 0.7071067811865475f));
}

// ---------------- Kw: one-time f32 -> f16 weight conversion ------------------

__global__ void k_cvt_weights(const float* __restrict__ gate_w, const float* __restrict__ nproj_w,
                              _Float16* __restrict__ gh, _Float16* __restrict__ nh) {
  int i = blockIdx.x * blockDim.x + threadIdx.x;
  if (i < 32768) gh[i] = (_Float16)gate_w[i];
  if (i < 16384) nh[i] = (_Float16)nproj_w[i];
}

// ---------------- K0: node = LayerNorm(transpose(feature_map)) ---------------

__global__ void k_node_ln(const float* __restrict__ fm, const float* __restrict__ g,
                          const float* __restrict__ bt, float* __restrict__ node) {
  int b = blockIdx.x / N_;
  int n = blockIdx.x % N_;
  int c = threadIdx.x;  // 128 threads
  float v = fm[(size_t)b * C_ * N_ + (size_t)c * N_ + n];
  __shared__ float red[128];
  __shared__ float mu_s, rs_s;
  red[c] = v;
  __syncthreads();
  for (int s = 64; s > 0; s >>= 1) { if (c < s) red[c] += red[c + s]; __syncthreads(); }
  if (c == 0) mu_s = red[0] * (1.0f / 128.0f);
  __syncthreads();
  float d = v - mu_s;
  red[c] = d * d;
  __syncthreads();
  for (int s = 64; s > 0; s >>= 1) { if (c < s) red[c] += red[c + s]; __syncthreads(); }
  if (c == 0) rs_s = rsqrtf(red[0] * (1.0f / 128.0f) + 1e-5f);
  __syncthreads();
  node[(size_t)b * N_ * C_ + (size_t)n * C_ + c] = d * rs_s * g[c] + bt[c];
}

// ---------------- K1a/K1b: hyper = S^T @ node (two-stage deterministic) ------

__global__ void k_hyper_partial(const float* __restrict__ node, const float* __restrict__ scores,
                                float* __restrict__ part) {
  int b = blockIdx.x / NCH, ch = blockIdx.x % NCH;
  int nbase = ch * 128;
  __shared__ float s_sc[128 * 16];
  int t = threadIdx.x;
  for (int i = t; i < 128 * 16; i += 256)
    s_sc[i] = scores[(size_t)b * N_ * E_ + (size_t)(nbase + (i >> 4)) * E_ + (i & 15)];
  __syncthreads();
  int c = t & 127, eh = t >> 7;
  float acc[8];
#pragma unroll
  for (int j = 0; j < 8; ++j) acc[j] = 0.0f;
  for (int n = 0; n < 128; ++n) {
    float x = node[(size_t)b * N_ * C_ + (size_t)(nbase + n) * C_ + c];
#pragma unroll
    for (int j = 0; j < 8; ++j) acc[j] += s_sc[n * 16 + eh * 8 + j] * x;
  }
  float* dst = part + (size_t)(b * NCH + ch) * 2048;
#pragma unroll
  for (int j = 0; j < 8; ++j) dst[(eh * 8 + j) * 128 + c] = acc[j];
}

__global__ void k_hyper_reduce(const float* __restrict__ part, float* __restrict__ hyper) {
  int b = blockIdx.x, t = threadIdx.x;
  for (int i = t; i < 2048; i += 256) {
    float s = 0.0f;
    for (int ch = 0; ch < NCH; ++ch) s += part[(size_t)(b * NCH + ch) * 2048 + i];
    hyper[(size_t)b * 2048 + i] = s;
  }
}

// ---------------- LN helper over 16 rows x `cols` (block-wide) ---------------

static __device__ __forceinline__ void ln_rows(const float* x, float* out, int cols,
                                               const float* g, const float* bt,
                                               float* s_mu, float* s_rs) {
  int t = threadIdx.x;
  if (t < 16) {
    float s = 0.0f;
    for (int c = 0; c < cols; ++c) s += x[t * cols + c];
    float mu = s / (float)cols;
    float q = 0.0f;
    for (int c = 0; c < cols; ++c) { float d = x[t * cols + c] - mu; q += d * d; }
    s_mu[t] = mu;
    s_rs[t] = rsqrtf(q / (float)cols + 1e-5f);
  }
  __syncthreads();
  int tot = 16 * cols;
  for (int i = t; i < tot; i += (int)blockDim.x) {
    int m = i / cols, c = i - m * cols;
    out[i] = (x[i] - s_mu[m]) * s_rs[m] * g[c] + bt[c];
  }
  __syncthreads();
}

// ---------------- K2: per-batch hyperedge transformer conv (tiny, VALU) ------
// Emits hyperP (f16), V (f16), KQ = K@q_w (f16) and qbK = K@q_b (f32):
// the node-side attention logits become node @ KQ^T + qbK, so the big
// per-node Q projection is folded away algebraically.

__global__ void k_edge_transformer(
    const float* __restrict__ hyper,
    const float* __restrict__ w_in, const float* __restrict__ b_in,
    const float* __restrict__ w_out, const float* __restrict__ b_out,
    const float* __restrict__ ffn_w1, const float* __restrict__ ffn_b1,
    const float* __restrict__ ffn_w2, const float* __restrict__ ffn_b2,
    const float* __restrict__ tn_g, const float* __restrict__ tn_b,
    const float* __restrict__ nh_g, const float* __restrict__ nh_b,
    const float* __restrict__ hp_w, const float* __restrict__ hp_b,
    const float* __restrict__ kv_w, const float* __restrict__ kv_b,
    const float* __restrict__ q_w, const float* __restrict__ q_b,
    _Float16* __restrict__ hyperP, _Float16* __restrict__ Vw,
    _Float16* __restrict__ KQw, float* __restrict__ qbKw) {
  __shared__ float s_h[2048], s_q[2048], s_k[2048], s_v[2048], s_x[2048];
  __shared__ float s_f[4096];
  __shared__ float s_att[1024];
  __shared__ float s_mu[16], s_rs[16];
  int b = blockIdx.x, t = threadIdx.x;
  for (int i = t; i < 2048; i += 256) s_h[i] = hyper[(size_t)b * 2048 + i];
  __syncthreads();
  // qkv projections
  for (int i = t; i < 2048; i += 256) {
    int m = i >> 7, c = i & 127;
    float aq = b_in[c], ak = b_in[128 + c], av = b_in[256 + c];
    for (int k = 0; k < 128; ++k) {
      float x = s_h[m * 128 + k];
      aq += x * w_in[c * 128 + k];
      ak += x * w_in[(128 + c) * 128 + k];
      av += x * w_in[(256 + c) * 128 + k];
    }
    s_q[i] = aq; s_k[i] = ak; s_v[i] = av;
  }
  __syncthreads();
  // attention logits (4 heads, hd=32)
  for (int i = t; i < 1024; i += 256) {
    int h = i >> 8, rem = i & 255, qi = rem >> 4, kj = rem & 15;
    float s = 0.0f;
    int off = h * 32;
    for (int d = 0; d < 32; ++d) s += s_q[qi * 128 + off + d] * s_k[kj * 128 + off + d];
    s_att[i] = s * 0.17677669529663687f;  // 1/sqrt(32)
  }
  __syncthreads();
  if (t < 64) {
    float* row = s_att + (t >> 4) * 256 + (t & 15) * 16;
    float mx = row[0];
    for (int j = 1; j < 16; ++j) mx = fmaxf(mx, row[j]);
    float sm = 0.0f;
    for (int j = 0; j < 16; ++j) { row[j] = __expf(row[j] - mx); sm += row[j]; }
    float inv = 1.0f / sm;
    for (int j = 0; j < 16; ++j) row[j] *= inv;
  }
  __syncthreads();
  // attn @ V -> s_x
  for (int i = t; i < 2048; i += 256) {
    int m = i >> 7, c = i & 127, h = c >> 5;
    float s = 0.0f;
    for (int j = 0; j < 16; ++j) s += s_att[h * 256 + m * 16 + j] * s_v[j * 128 + c];
    s_x[i] = s;
  }
  __syncthreads();
  // out-proj + residual -> s_q
  for (int i = t; i < 2048; i += 256) {
    int m = i >> 7, c = i & 127;
    float s = b_out[c];
    for (int k = 0; k < 128; ++k) s += s_x[m * 128 + k] * w_out[c * 128 + k];
    s_q[i] = s_h[i] + s;
  }
  __syncthreads();
  ln_rows(s_q, s_x, 128, tn_g, tn_b, s_mu, s_rs);      // x
  // FFN hidden (gelu)
  for (int i = t; i < 4096; i += 256) {
    int m = i >> 8, j = i & 255;
    float s = ffn_b1[j];
    for (int k = 0; k < 128; ++k) s += s_x[m * 128 + k] * ffn_w1[j * 128 + k];
    s_f[i] = gelu_f(s);
  }
  __syncthreads();
  // FFN out + residual -> s_q
  for (int i = t; i < 2048; i += 256) {
    int m = i >> 7, c = i & 127;
    float s = ffn_b2[c];
    for (int j = 0; j < 256; ++j) s += s_f[m * 256 + j] * ffn_w2[c * 256 + j];
    s_q[i] = s_x[i] + s;
  }
  __syncthreads();
  ln_rows(s_q, s_q, 128, tn_g, tn_b, s_mu, s_rs);      // hyper (post conv)
  ln_rows(s_q, s_x, 128, nh_g, nh_b, s_mu, s_rs);      // norm_hyper
  // hyper_proj + gelu -> s_h + ws (f16)
  for (int i = t; i < 2048; i += 256) {
    int m = i >> 7, c = i & 127;
    float s = hp_b[c];
    for (int k = 0; k < 128; ++k) s += s_x[m * 128 + k] * hp_w[c * 128 + k];
    float v = gelu_f(s);
    s_h[i] = v;
    hyperP[(size_t)b * 2048 + i] = (_Float16)v;
  }
  __syncthreads();
  // K,V projections: K kept in LDS (s_k), V straight to ws (f16)
  for (int i = t; i < 2048; i += 256) {
    int m = i >> 7, c = i & 127;
    float sk = kv_b[c], sv = kv_b[128 + c];
    for (int k = 0; k < 128; ++k) {
      float x = s_h[m * 128 + k];
      sk += x * kv_w[c * 128 + k];
      sv += x * kv_w[(128 + c) * 128 + k];
    }
    s_k[i] = sk;
    Vw[(size_t)b * 2048 + i] = (_Float16)sv;
  }
  __syncthreads();
  // KQ = K @ q_w  (q_w indexed [c_out*128 + c_in]; KQ[e,k] = sum_c K[e,c]*q_w[c,k])
  for (int i = t; i < 2048; i += 256) {
    int e = i >> 7, k = i & 127;
    float s = 0.0f;
    for (int c = 0; c < 128; ++c) s += s_k[e * 128 + c] * q_w[c * 128 + k];
    KQw[(size_t)b * 2048 + i] = (_Float16)s;
  }
  // qbK[e] = sum_c q_b[c] * K[e,c]
  if (t < 16) {
    float s = 0.0f;
    for (int c = 0; c < 128; ++c) s += q_b[c] * s_k[t * 128 + c];
    qbKw[b * 16 + t] = s;
  }
}

// ---------------- K_main: fused node-side pipeline (WMMA) --------------------

__global__ void __launch_bounds__(256)
k_node_fused(const float* __restrict__ scores, const float* __restrict__ node,
             const _Float16* __restrict__ hyperP, const _Float16* __restrict__ KQw,
             const _Float16* __restrict__ Vw, const float* __restrict__ qbKw,
             const float* __restrict__ nf_g, const float* __restrict__ nf_b,
             const _Float16* __restrict__ gate_wh, const float* __restrict__ gate_b,
             const float* __restrict__ gl_g, const float* __restrict__ gl_b,
             const _Float16* __restrict__ nproj_wh, const float* __restrict__ nproj_b,
             float* __restrict__ out) {
  __shared__ float    s_node[16 * 128];
  __shared__ _Float16 s_node_h[16 * 128];
  __shared__ float    s_sc[16 * 16];
  __shared__ _Float16 s_hp[16 * 128];
  __shared__ _Float16 s_KQ[16 * 128];
  __shared__ _Float16 s_V[16 * 128];
  __shared__ float    s_a[16 * 16];
  __shared__ float    s_qbk[16];
  __shared__ float    s_comb[16 * 256];
  __shared__ _Float16 s_combn[16 * 256];
  __shared__ float    s_gate[16 * 128];
  __shared__ float    s_mu[16], s_rs[16];

  int b = blockIdx.x / NTILES;
  int n0 = (blockIdx.x % NTILES) * 16;
  int t = threadIdx.x, wave = t >> 5, lane = t & 31;
  int wcol = wave * 16;

  // cooperative loads (hp/KQ/V already f16 in ws; node gets an f16 shadow)
  const float* nodeT = node + (size_t)b * N_ * C_ + (size_t)n0 * C_;
  for (int i = t; i < 2048; i += 256) {
    float nv = nodeT[i];
    s_node[i] = nv;
    s_node_h[i] = (_Float16)nv;
    s_hp[i] = hyperP[(size_t)b * 2048 + i];
    s_KQ[i] = KQw[(size_t)b * 2048 + i];
    s_V[i]  = Vw[(size_t)b * 2048 + i];
  }
  s_sc[t & 255] = scores[(size_t)b * N_ * E_ + (size_t)(n0 + (t >> 4)) * E_ + (t & 15)];
  if (t < 16) s_qbk[t] = qbKw[b * 16 + t];
  __syncthreads();

  // logits = node @ KQ^T + qbK, scaled, softmax over E (wave 0 only).
  // Dual accumulator chains to hide WMMA->WMMA RAW spacing.
  if (wave == 0) {
    v8f acc0 = {}, acc1 = {};
    acc0 = wmma_f16(load_A_h(s_node_h +  0, 128, lane), load_B_wT_h(s_KQ, 128, 0,  0, lane), acc0);
    acc1 = wmma_f16(load_A_h(s_node_h + 32, 128, lane), load_B_wT_h(s_KQ, 128, 0, 32, lane), acc1);
    acc0 = wmma_f16(load_A_h(s_node_h + 64, 128, lane), load_B_wT_h(s_KQ, 128, 0, 64, lane), acc0);
    acc1 = wmma_f16(load_A_h(s_node_h + 96, 128, lane), load_B_wT_h(s_KQ, 128, 0, 96, lane), acc1);
    const float scale = 0.08838834764831845f;  // 1/sqrt(128)
    int mb = (lane >> 4) * 8, col = lane & 15;
    float qb = s_qbk[col];
#pragma unroll
    for (int r = 0; r < 8; ++r) {
      float v = (acc0[r] + acc1[r] + qb) * scale;
      float mx = v;
      for (int m = 8; m >= 1; m >>= 1) mx = fmaxf(mx, __shfl_xor(mx, m, 32));
      float e = __expf(v - mx);
      float sm = e;
      for (int m = 8; m >= 1; m >>= 1) sm += __shfl_xor(sm, m, 32);
      s_a[(mb + r) * 16 + col] = e / sm;
    }
  }
  __syncthreads();

  // intra = scores_tile @ hyperP ; cross = a @ V  (K=16, zero-padded to 32)
  {
    v8f acc_i = {}, acc_x = {};
    v16h a_sc = load_A_pad16_f32(s_sc, 16, lane);
    v16h b_hp = load_B_rm_h(s_hp, 128, wcol, 16, lane);
    acc_i = wmma_f16(a_sc, b_hp, acc_i);
    v16h a_a = load_A_pad16_f32(s_a, 16, lane);
    v16h b_v = load_B_rm_h(s_V, 128, wcol, 16, lane);
    acc_x = wmma_f16(a_a, b_v, acc_x);
    int col = wcol + (lane & 15);
    int mb = (lane >> 4) * 8;
#pragma unroll
    for (int r = 0; r < 8; ++r) {
      s_comb[(mb + r) * 256 + col]       = acc_i[r];
      s_comb[(mb + r) * 256 + 128 + col] = acc_x[r];
    }
  }
  __syncthreads();

  // norm_fusion over 2C=256, write f16 for gate GEMM
  if (t < 16) {
    float s = 0.0f;
    for (int c = 0; c < 256; ++c) s += s_comb[t * 256 + c];
    float mu = s * (1.0f / 256.0f);
    float q = 0.0f;
    for (int c = 0; c < 256; ++c) { float d = s_comb[t * 256 + c] - mu; q += d * d; }
    s_mu[t] = mu;
    s_rs[t] = rsqrtf(q * (1.0f / 256.0f) + 1e-5f);
  }
  __syncthreads();
  for (int i = t; i < 4096; i += 256) {
    int m = i >> 8, c = i & 255;
    s_combn[i] = (_Float16)((s_comb[i] - s_mu[m]) * s_rs[m] * nf_g[c] + nf_b[c]);
  }
  __syncthreads();

  // gate_lin = combined_n @ gate_w^T + gate_b  (K=256: dual chains of 4 WMMAs)
  {
    int col = wcol + (lane & 15);
    float bias = gate_b[col];
    v8f acc0, acc1 = {};
#pragma unroll
    for (int r = 0; r < 8; ++r) acc0[r] = bias;
#pragma unroll
    for (int kk = 0; kk < 256; kk += 64) {
      acc0 = wmma_f16(load_A_h(s_combn + kk,      256, lane), load_B_wT_h(gate_wh, 256, wcol, kk,      lane), acc0);
      acc1 = wmma_f16(load_A_h(s_combn + kk + 32, 256, lane), load_B_wT_h(gate_wh, 256, wcol, kk + 32, lane), acc1);
    }
    int mb = (lane >> 4) * 8;
#pragma unroll
    for (int r = 0; r < 8; ++r) s_gate[(mb + r) * 128 + col] = acc0[r] + acc1[r];
  }
  __syncthreads();

  // gate = sigmoid(LN(gate_lin))
  if (t < 16) {
    float s = 0.0f;
    for (int c = 0; c < 128; ++c) s += s_gate[t * 128 + c];
    float mu = s * (1.0f / 128.0f);
    float q = 0.0f;
    for (int c = 0; c < 128; ++c) { float d = s_gate[t * 128 + c] - mu; q += d * d; }
    s_mu[t] = mu;
    s_rs[t] = rsqrtf(q * (1.0f / 128.0f) + 1e-5f);
  }
  __syncthreads();
  for (int i = t; i < 2048; i += 256) {
    int m = i >> 7, c = i & 127;
    float v = (s_gate[i] - s_mu[m]) * s_rs[m] * gl_g[c] + gl_b[c];
    s_gate[i] = 1.0f / (1.0f + __expf(-v));
  }
  __syncthreads();

  // upd = node + gate*intra + (1-gate)*cross  (f32 + refresh f16 shadow)
  for (int i = t; i < 2048; i += 256) {
    int m = i >> 7, c = i & 127;
    float g = s_gate[i];
    float intra = s_comb[m * 256 + c];
    float cross = s_comb[m * 256 + 128 + c];
    float upd = s_node[i] + g * intra + (1.0f - g) * cross;
    s_node[i] = upd;
    s_node_h[i] = (_Float16)upd;
  }
  __syncthreads();

  // out = gelu(upd @ nproj_w^T + nproj_b), stored transposed [B,C,N]
  {
    int col = wcol + (lane & 15);
    float bias = nproj_b[col];
    v8f acc0, acc1 = {};
#pragma unroll
    for (int r = 0; r < 8; ++r) acc0[r] = bias;
    acc0 = wmma_f16(load_A_h(s_node_h +  0, 128, lane), load_B_wT_h(nproj_wh, 128, wcol,  0, lane), acc0);
    acc1 = wmma_f16(load_A_h(s_node_h + 32, 128, lane), load_B_wT_h(nproj_wh, 128, wcol, 32, lane), acc1);
    acc0 = wmma_f16(load_A_h(s_node_h + 64, 128, lane), load_B_wT_h(nproj_wh, 128, wcol, 64, lane), acc0);
    acc1 = wmma_f16(load_A_h(s_node_h + 96, 128, lane), load_B_wT_h(nproj_wh, 128, wcol, 96, lane), acc1);
    int mb = (lane >> 4) * 8;
#pragma unroll
    for (int r = 0; r < 8; ++r) {
      float v = gelu_f(acc0[r] + acc1[r]);
      out[(size_t)b * C_ * N_ + (size_t)col * N_ + n0 + mb + r] = v;
    }
  }
}

// ---------------- launch ------------------------------------------------------

extern "C" void kernel_launch(void* const* d_in, const int* in_sizes, int n_in,
                              void* d_out, int out_size, void* d_ws, size_t ws_size,
                              hipStream_t stream) {
  (void)in_sizes; (void)n_in; (void)out_size; (void)ws_size;
  const float* fm      = (const float*)d_in[0];
  const float* scores  = (const float*)d_in[1];
  const float* np_g    = (const float*)d_in[2];
  const float* np_b    = (const float*)d_in[3];
  const float* mha_w_in  = (const float*)d_in[4];
  const float* mha_b_in  = (const float*)d_in[5];
  const float* mha_w_out = (const float*)d_in[6];
  const float* mha_b_out = (const float*)d_in[7];
  const float* ffn_w1  = (const float*)d_in[8];
  const float* ffn_b1  = (const float*)d_in[9];
  const float* ffn_w2  = (const float*)d_in[10];
  const float* ffn_b2  = (const float*)d_in[11];
  const float* tn_g    = (const float*)d_in[12];
  const float* tn_b    = (const float*)d_in[13];
  const float* nh_g    = (const float*)d_in[14];
  const float* nh_b    = (const float*)d_in[15];
  const float* hp_w    = (const float*)d_in[16];
  const float* hp_b    = (const float*)d_in[17];
  const float* q_w     = (const float*)d_in[18];
  const float* q_b     = (const float*)d_in[19];
  const float* kv_w    = (const float*)d_in[20];
  const float* kv_b    = (const float*)d_in[21];
  const float* nf_g    = (const float*)d_in[22];
  const float* nf_b    = (const float*)d_in[23];
  const float* gate_w  = (const float*)d_in[24];
  const float* gate_b  = (const float*)d_in[25];
  const float* gl_g    = (const float*)d_in[26];
  const float* gl_b    = (const float*)d_in[27];
  const float* nproj_w = (const float*)d_in[28];
  const float* nproj_b = (const float*)d_in[29];
  float* out = (float*)d_out;

  float* ws = (float*)d_ws;
  const size_t SZ_NODE = (size_t)B_ * N_ * C_;       // 9,437,184 f32
  const size_t SZ_PART = (size_t)B_ * NCH * 2048;    // 1,179,648 f32
  float* ws_node  = ws;
  float* ws_part  = ws_node + SZ_NODE;
  float* ws_hyper = ws_part + SZ_PART;
  float* ws_qbK   = ws_hyper + (size_t)B_ * 2048;    // B*16 f32
  float* ws_end_f = ws_qbK + (size_t)B_ * 16;
  _Float16* hbase   = (_Float16*)ws_end_f;
  _Float16* ws_hpH  = hbase;                          // B*2048 f16
  _Float16* ws_VH   = ws_hpH + (size_t)B_ * 2048;
  _Float16* ws_KQH  = ws_VH  + (size_t)B_ * 2048;
  _Float16* ws_gwH  = ws_KQH + (size_t)B_ * 2048;     // 32768
  _Float16* ws_npH  = ws_gwH + 32768;                 // 16384

  k_cvt_weights<<<128, 256, 0, stream>>>(gate_w, nproj_w, ws_gwH, ws_npH);
  k_node_ln<<<B_ * N_, 128, 0, stream>>>(fm, np_g, np_b, ws_node);
  k_hyper_partial<<<B_ * NCH, 256, 0, stream>>>(ws_node, scores, ws_part);
  k_hyper_reduce<<<B_, 256, 0, stream>>>(ws_part, ws_hyper);
  k_edge_transformer<<<B_, 256, 0, stream>>>(
      ws_hyper, mha_w_in, mha_b_in, mha_w_out, mha_b_out,
      ffn_w1, ffn_b1, ffn_w2, ffn_b2, tn_g, tn_b, nh_g, nh_b,
      hp_w, hp_b, kv_w, kv_b, q_w, q_b, ws_hpH, ws_VH, ws_KQH, ws_qbK);
  k_node_fused<<<B_ * NTILES, 256, 0, stream>>>(
      scores, ws_node, ws_hpH, ws_KQH, ws_VH, ws_qbK,
      nf_g, nf_b, ws_gwH, gate_b, gl_g, gl_b,
      ws_npH, nproj_b, out);
}
